// GCNWithLLMFeature_25649544691873
// MI455X (gfx1250) — matrix operand order repaired and checked by reading, not compiled, not run on hardware
//
#include <hip/hip_runtime.h>
#include <hip/hip_bf16.h>

typedef __attribute__((ext_vector_type(16))) __bf16 v16bf;
typedef __attribute__((ext_vector_type(8)))  __bf16 v8bf;
typedef __attribute__((ext_vector_type(8)))  float  v8f;

#define WMMA_BF16(a, b, c) \
  __builtin_amdgcn_wmma_f32_16x16x32_bf16(false, (a), false, (b), (short)0, (c), false, false)

static __device__ __forceinline__ v16bf cat8(v8bf lo, v8bf hi) {
  return __builtin_shufflevector(lo, hi, 0, 1, 2, 3, 4, 5, 6, 7, 8, 9, 10, 11, 12, 13, 14, 15);
}

static __device__ __forceinline__ v8bf cvt8(float4 a, float4 b) {
  v8bf r;
  r[0] = (__bf16)a.x; r[1] = (__bf16)a.y; r[2] = (__bf16)a.z; r[3] = (__bf16)a.w;
  r[4] = (__bf16)b.x; r[5] = (__bf16)b.y; r[6] = (__bf16)b.z; r[7] = (__bf16)b.w;
  return r;
}

static __device__ __forceinline__ v8bf cvt8_relu(float4 a, float4 b) {
  v8bf r;
  r[0] = (__bf16)fmaxf(a.x, 0.f); r[1] = (__bf16)fmaxf(a.y, 0.f);
  r[2] = (__bf16)fmaxf(a.z, 0.f); r[3] = (__bf16)fmaxf(a.w, 0.f);
  r[4] = (__bf16)fmaxf(b.x, 0.f); r[5] = (__bf16)fmaxf(b.y, 0.f);
  r[6] = (__bf16)fmaxf(b.z, 0.f); r[7] = (__bf16)fmaxf(b.w, 0.f);
  return r;
}

static __device__ __forceinline__ v16bf loadB16(const __bf16* p) {
  return cat8(*(const v8bf*)p, *(const v8bf*)(p + 16));
}

// ---------------------------------------------------------------- utilities
__global__ void k_zero(float* __restrict__ p, int n) {
  int i = blockIdx.x * blockDim.x + threadIdx.x;
  if (i < n) p[i] = 0.0f;
}

// src is [K x Ncols] f32 row-major; dst is [Ncols x Kpad] bf16 (transposed, K-padded w/ 0)
__global__ void k_prep_wt(const float* __restrict__ src, __bf16* __restrict__ dst,
                          int K, int Kpad, int Ncols, int total) {
  int id = blockIdx.x * blockDim.x + threadIdx.x;
  if (id >= total) return;
  int n = id / Kpad, k = id - n * Kpad;
  float v = (k < K) ? src[(size_t)k * Ncols + n] : 0.0f;
  dst[(size_t)n * Kpad + k] = (__bf16)v;
}

__global__ void k_deg(const int* __restrict__ col, const float* __restrict__ ew,
                      float* __restrict__ deg, int E) {
  int e = blockIdx.x * blockDim.x + threadIdx.x;
  if (e < E) atomicAdd(&deg[col[e]], ew[e]);
}

__global__ void k_dinv(const float* __restrict__ deg, float* __restrict__ dinv, int N) {
  int n = blockIdx.x * blockDim.x + threadIdx.x;
  if (n < N) dinv[n] = rsqrtf(deg[n] + 1.0f);  // +1 self loop => deg > 0 always
}

__global__ void k_normw(const int* __restrict__ row, const int* __restrict__ col,
                        const float* __restrict__ ew, const float* __restrict__ dinv,
                        float* __restrict__ nrm, int E) {
  int e = blockIdx.x * blockDim.x + threadIdx.x;
  if (e < E) nrm[e] = dinv[row[e]] * ew[e] * dinv[col[e]];
}

// ---------------------------------------------------- feature build + LLM proj (WMMA)
// xb layout per node (bf16, 160 cols): [0,64) name | [64,80) type | [80,112) llm_proj
// | [112,144) behaviors | [144,160) zero pad
__global__ void __launch_bounds__(32) k_features(
    const int* __restrict__ xnames, const int* __restrict__ xtypes,
    const float* __restrict__ xbeh, const float* __restrict__ llm,
    const float* __restrict__ name_emb, const float* __restrict__ type_emb,
    const __bf16* __restrict__ btllm, const float* __restrict__ lpb,
    __bf16* __restrict__ xb) {
  const int lane = threadIdx.x & 31;
  const int g = lane >> 4, r = lane & 15;
  const int tile = blockIdx.x;
  const int rowA = tile * 16 + r;
  const float* Ar = llm + (size_t)rowA * 768;

  v8f c0 = {}, c1 = {};
#pragma unroll 4
  for (int k0 = 0; k0 < 768; k0 += 32) {
    float4 a0 = *(const float4*)(Ar + k0 + 8 * g);
    float4 a1 = *(const float4*)(Ar + k0 + 8 * g + 4);
    float4 a2 = *(const float4*)(Ar + k0 + 16 + 8 * g);
    float4 a3 = *(const float4*)(Ar + k0 + 16 + 8 * g + 4);
    v16bf a = cat8(cvt8(a0, a1), cvt8(a2, a3));
    const __bf16* bp0 = btllm + (size_t)r * 768 + k0 + 8 * g;
    const __bf16* bp1 = btllm + (size_t)(16 + r) * 768 + k0 + 8 * g;
    c0 = WMMA_BF16(a, loadB16(bp0), c0);
    c1 = WMMA_BF16(a, loadB16(bp1), c1);
  }
  const float pb0 = lpb[r], pb1 = lpb[16 + r];
#pragma unroll
  for (int vr = 0; vr < 8; ++vr) {
    size_t m = (size_t)(tile * 16 + vr + 8 * g);
    xb[m * 160 + 80 + r] = (__bf16)(c0[vr] + pb0);
    xb[m * 160 + 96 + r] = (__bf16)(c1[vr] + pb1);
  }

  // gathers: lane handles row rowA, column-half g
  __bf16* xr = xb + (size_t)rowA * 160;
  const float* ne = name_emb + (size_t)xnames[rowA] * 64 + 32 * g;
#pragma unroll
  for (int i = 0; i < 32; i += 8) {
    float4 u = *(const float4*)(ne + i), v = *(const float4*)(ne + i + 4);
    *(v8bf*)(xr + 32 * g + i) = cvt8(u, v);
  }
  const float* te = type_emb + (size_t)xtypes[rowA] * 16 + 8 * g;
  {
    float4 u = *(const float4*)(te), v = *(const float4*)(te + 4);
    *(v8bf*)(xr + 64 + 8 * g) = cvt8(u, v);
  }
  const float* be = xbeh + (size_t)rowA * 32 + 16 * g;
#pragma unroll
  for (int i = 0; i < 16; i += 8) {
    float4 u = *(const float4*)(be + i), v = *(const float4*)(be + i + 4);
    *(v8bf*)(xr + 112 + 16 * g + i) = cvt8(u, v);
  }
  v8bf z = {};
  *(v8bf*)(xr + 144 + 8 * g) = z;
}

// ------------------------------------------- GEMM: [Nx Kpad] bf16 @ [Kpad x 64] -> f32
__global__ void __launch_bounds__(32) k_gemm_bf16(
    const __bf16* __restrict__ A, const __bf16* __restrict__ Bt,
    float* __restrict__ out, int Kpad) {
  const int lane = threadIdx.x & 31;
  const int g = lane >> 4, r = lane & 15;
  const int tile = blockIdx.x;
  const int rowA = tile * 16 + r;
  const __bf16* Ar = A + (size_t)rowA * Kpad;

  v8f acc0 = {}, acc1 = {}, acc2 = {}, acc3 = {};
  for (int k0 = 0; k0 < Kpad; k0 += 32) {
    v16bf a = cat8(*(const v8bf*)(Ar + k0 + 8 * g), *(const v8bf*)(Ar + k0 + 16 + 8 * g));
    const __bf16* b0 = Bt + (size_t)(r)*Kpad + k0 + 8 * g;
    const __bf16* b1 = Bt + (size_t)(16 + r) * Kpad + k0 + 8 * g;
    const __bf16* b2 = Bt + (size_t)(32 + r) * Kpad + k0 + 8 * g;
    const __bf16* b3 = Bt + (size_t)(48 + r) * Kpad + k0 + 8 * g;
    acc0 = WMMA_BF16(a, loadB16(b0), acc0);
    acc1 = WMMA_BF16(a, loadB16(b1), acc1);
    acc2 = WMMA_BF16(a, loadB16(b2), acc2);
    acc3 = WMMA_BF16(a, loadB16(b3), acc3);
  }
#pragma unroll
  for (int vr = 0; vr < 8; ++vr) {
    size_t m = (size_t)(tile * 16 + vr + 8 * g);
    out[m * 64 + r] = acc0[vr];
    out[m * 64 + 16 + r] = acc1[vr];
    out[m * 64 + 32 + r] = acc2[vr];
    out[m * 64 + 48 + r] = acc3[vr];
  }
}

// ----------------------------- GEMM with inline ReLU+cvt on f32 A: [Nx64] @ [64x64]
__global__ void __launch_bounds__(32) k_gemm_relu(
    const float* __restrict__ A, const __bf16* __restrict__ Bt,
    float* __restrict__ out) {
  const int lane = threadIdx.x & 31;
  const int g = lane >> 4, r = lane & 15;
  const int tile = blockIdx.x;
  const int rowA = tile * 16 + r;
  const float* Ar = A + (size_t)rowA * 64;

  v8f acc0 = {}, acc1 = {}, acc2 = {}, acc3 = {};
#pragma unroll
  for (int k0 = 0; k0 < 64; k0 += 32) {
    float4 f0 = *(const float4*)(Ar + k0 + 8 * g);
    float4 f1 = *(const float4*)(Ar + k0 + 8 * g + 4);
    float4 f2 = *(const float4*)(Ar + k0 + 16 + 8 * g);
    float4 f3 = *(const float4*)(Ar + k0 + 16 + 8 * g + 4);
    v16bf a = cat8(cvt8_relu(f0, f1), cvt8_relu(f2, f3));
    const __bf16* b0 = Bt + (size_t)(r)*64 + k0 + 8 * g;
    const __bf16* b1 = Bt + (size_t)(16 + r) * 64 + k0 + 8 * g;
    const __bf16* b2 = Bt + (size_t)(32 + r) * 64 + k0 + 8 * g;
    const __bf16* b3 = Bt + (size_t)(48 + r) * 64 + k0 + 8 * g;
    acc0 = WMMA_BF16(a, loadB16(b0), acc0);
    acc1 = WMMA_BF16(a, loadB16(b1), acc1);
    acc2 = WMMA_BF16(a, loadB16(b2), acc2);
    acc3 = WMMA_BF16(a, loadB16(b3), acc3);
  }
#pragma unroll
  for (int vr = 0; vr < 8; ++vr) {
    size_t m = (size_t)(tile * 16 + vr + 8 * g);
    out[m * 64 + r] = acc0[vr];
    out[m * 64 + 16 + r] = acc1[vr];
    out[m * 64 + 32 + r] = acc2[vr];
    out[m * 64 + 48 + r] = acc3[vr];
  }
}

// ---------------------------------------------------------------- GCN aggregation
// out[n][c] = dinv[n]^2 * hlin[n][c] + bias[c]   (self-loop term + bias)
__global__ void k_init_out(const float* __restrict__ hlin, const float* __restrict__ dinv,
                           const float* __restrict__ bias, float* __restrict__ out, int N) {
  int id = blockIdx.x * blockDim.x + threadIdx.x;
  if (id >= N * 64) return;
  int n = id >> 6, c = id & 63;
  float s = dinv[n];
  out[id] = s * s * hlin[id] + bias[c];
}

// 32 lanes per edge, 2 channels per lane
__global__ void k_scatter(const int* __restrict__ row, const int* __restrict__ col,
                          const float* __restrict__ nrm, const float* __restrict__ hlin,
                          float* __restrict__ out, int E) {
  int gid = blockIdx.x * blockDim.x + threadIdx.x;
  int e = gid >> 5;
  if (e >= E) return;
  int lane = gid & 31;
  int rr = row[e], cc = col[e];
  float w = nrm[e];
  atomicAdd(&out[(size_t)cc * 64 + lane], w * hlin[(size_t)rr * 64 + lane]);
  atomicAdd(&out[(size_t)cc * 64 + lane + 32], w * hlin[(size_t)rr * 64 + lane + 32]);
}

// ---------------------------------------------------------------- pooling + classifier
__global__ void k_pool(const float* __restrict__ h, const int* __restrict__ batch,
                       float* __restrict__ psum, int N) {
  int gid = blockIdx.x * blockDim.x + threadIdx.x;
  int n = gid >> 5;
  if (n >= N) return;
  int lane = gid & 31;
  int gph = batch[n];
  atomicAdd(&psum[gph * 64 + lane], fmaxf(h[(size_t)n * 64 + lane], 0.0f));
  atomicAdd(&psum[gph * 64 + lane + 32], fmaxf(h[(size_t)n * 64 + lane + 32], 0.0f));
}

__global__ void k_cnt(const int* __restrict__ batch, float* __restrict__ cnt, int N) {
  int n = blockIdx.x * blockDim.x + threadIdx.x;
  if (n < N) atomicAdd(&cnt[batch[n]], 1.0f);
}

__global__ void k_final(const float* __restrict__ psum, const float* __restrict__ cnt,
                        const float* __restrict__ clsw, const float* __restrict__ clsb,
                        float* __restrict__ out) {
  int t = threadIdx.x;
  if (t >= 256) return;
  int g = t >> 1, k = t & 1;
  float inv = 1.0f / fmaxf(cnt[g], 1.0f);
  float acc = 0.0f;
#pragma unroll
  for (int c = 0; c < 64; ++c) acc += psum[g * 64 + c] * clsw[c * 2 + k];
  out[g * 2 + k] = acc * inv + clsb[k];
}

// ---------------------------------------------------------------------------- launch
extern "C" void kernel_launch(void* const* d_in, const int* in_sizes, int n_in,
                              void* d_out, int out_size, void* d_ws, size_t ws_size,
                              hipStream_t stream) {
  const int* xnames = (const int*)d_in[0];
  const int* xtypes = (const int*)d_in[1];
  const float* xbeh = (const float*)d_in[2];
  const int* eidx = (const int*)d_in[3];
  const float* ew = (const float*)d_in[4];
  const int* batch = (const int*)d_in[5];
  const float* llm = (const float*)d_in[6];
  const float* name_emb = (const float*)d_in[7];
  const float* type_emb = (const float*)d_in[8];
  const float* lpw = (const float*)d_in[9];
  const float* lpb = (const float*)d_in[10];
  const float* w1 = (const float*)d_in[11];
  const float* b1 = (const float*)d_in[12];
  const float* w2 = (const float*)d_in[13];
  const float* b2 = (const float*)d_in[14];
  const float* clsw = (const float*)d_in[15];
  const float* clsb = (const float*)d_in[16];

  const int N = in_sizes[0];       // 100000 (divisible by 16)
  const int E = in_sizes[4];       // 1600000
  const int* erow = eidx;
  const int* ecol = eidx + E;

  size_t off = 0;
  auto carve = [&](size_t bytes) -> char* {
    char* p = (char*)d_ws + off;
    off += (bytes + 255) & ~(size_t)255;
    return p;
  };
  __bf16* xb = (__bf16*)carve((size_t)N * 160 * 2);   // node features, bf16, K-padded
  float* hA = (float*)carve((size_t)N * 64 * 4);      // h1lin -> h2lin
  float* hB = (float*)carve((size_t)N * 64 * 4);      // out1  -> out2
  __bf16* btll = (__bf16*)carve((size_t)32 * 768 * 2);
  __bf16* w1t = (__bf16*)carve((size_t)64 * 160 * 2);
  __bf16* w2t = (__bf16*)carve((size_t)64 * 64 * 2);
  float* deg = (float*)carve((size_t)N * 4);
  float* dinv = (float*)carve((size_t)N * 4);
  float* nrm = (float*)carve((size_t)E * 4);
  float* psum = (float*)carve((size_t)128 * 64 * 4);
  float* cnt = (float*)carve((size_t)128 * 4);

  const int tiles = N / 16;

  k_zero<<<(N + 255) / 256, 256, 0, stream>>>(deg, N);
  k_zero<<<(128 * 64 + 255) / 256, 256, 0, stream>>>(psum, 128 * 64);
  k_zero<<<1, 256, 0, stream>>>(cnt, 128);

  k_prep_wt<<<(32 * 768 + 255) / 256, 256, 0, stream>>>(lpw, btll, 768, 768, 32, 32 * 768);
  k_prep_wt<<<(64 * 160 + 255) / 256, 256, 0, stream>>>(w1, w1t, 144, 160, 64, 64 * 160);
  k_prep_wt<<<(64 * 64 + 255) / 256, 256, 0, stream>>>(w2, w2t, 64, 64, 64, 64 * 64);

  k_deg<<<(E + 255) / 256, 256, 0, stream>>>(ecol, ew, deg, E);
  k_dinv<<<(N + 255) / 256, 256, 0, stream>>>(deg, dinv, N);
  k_normw<<<(E + 255) / 256, 256, 0, stream>>>(erow, ecol, ew, dinv, nrm, E);

  k_features<<<tiles, 32, 0, stream>>>(xnames, xtypes, xbeh, llm, name_emb, type_emb,
                                       btll, lpb, xb);

  // conv1
  k_gemm_bf16<<<tiles, 32, 0, stream>>>(xb, w1t, hA, 160);
  k_init_out<<<(unsigned)(((size_t)N * 64 + 255) / 256), 256, 0, stream>>>(hA, dinv, b1, hB, N);
  k_scatter<<<(unsigned)(((size_t)E * 32 + 255) / 256), 256, 0, stream>>>(erow, ecol, nrm, hA, hB, E);

  // conv2 (ReLU of out1 folded into A conversion)
  k_gemm_relu<<<tiles, 32, 0, stream>>>(hB, w2t, hA);
  k_init_out<<<(unsigned)(((size_t)N * 64 + 255) / 256), 256, 0, stream>>>(hA, dinv, b2, hB, N);
  k_scatter<<<(unsigned)(((size_t)E * 32 + 255) / 256), 256, 0, stream>>>(erow, ecol, nrm, hA, hB, E);

  // pool (ReLU inside) + classifier
  k_pool<<<(unsigned)(((size_t)N * 32 + 255) / 256), 256, 0, stream>>>(hB, batch, psum, N);
  k_cnt<<<(N + 255) / 256, 256, 0, stream>>>(batch, cnt, N);
  k_final<<<1, 256, 0, stream>>>(psum, cnt, clsw, clsb, (float*)d_out);
}
